// MainEffectBlock_4973572129353
// MI455X (gfx1250) — matrix operand order repaired and checked by reading, not compiled, not run on hardware
//
#include <hip/hip_runtime.h>

typedef __attribute__((ext_vector_type(16))) _Float16 v16h;
typedef __attribute__((ext_vector_type(2)))  _Float16 h2v;
typedef __attribute__((ext_vector_type(8)))  float    v8f;

#define NN 16
#define NC 4
#define CATS 32
#define H1 40
#define H2 20
#define NF 20          // 16 numeric + 4 categorical features per row
#define TILE 256       // batch rows per block
#define THREADS 256    // 8 waves of 32

#define WMMA_F16(A, B, C) \
    __builtin_amdgcn_wmma_f32_16x16x32_f16(false, (A), false, (B), (short)0, (C), false, false)

struct BPack { h2v p[8]; };   // bit-castable to v16h (32 bytes)

__global__ __launch_bounds__(THREADS) void nam_forward_kernel(
    const float* __restrict__ inputs,   // [B, 20]
    const float* __restrict__ W1,       // [16, 1, 40]
    const float* __restrict__ b1,       // [16, 40]
    const float* __restrict__ W2,       // [16, 40, 20]
    const float* __restrict__ b2,       // [16, 20]
    const float* __restrict__ Wout,     // [16, 20]
    const float* __restrict__ bout,     // [16]
    const float* __restrict__ minv,     // [16]
    const float* __restrict__ maxv,     // [16]
    const float* __restrict__ cemb,     // [4, 32]
    const float* __restrict__ cobias,   // [4]
    float* __restrict__ out)            // [B, 20]
{
    // ---- LDS ------------------------------------------------------------
    __shared__ _Float16 sW2h[NN * H1 * H2];
    __shared__ float sW1[NN * H1], sB1[NN * H1];
    __shared__ float sB2[NN * H2], sWout[NN * H2];
    __shared__ float sBout[NN], sMin[NN], sMax[NN];
    __shared__ float sEmb[NC * CATS], sObias[NC];
    __shared__ __align__(16) float sX[TILE * NF];    // input tile -> output tile

    const int tid = threadIdx.x;
    const size_t rowBase = (size_t)blockIdx.x * TILE;

    // ---- async DMA of the input tile straight into LDS (ASYNCcnt) -------
    // Each thread moves 16B per round; 5 rounds cover 256x20 f32 = 20 KB.
    {
        const uint32_t sxOff = (uint32_t)(uintptr_t)(&sX[0]) + (uint32_t)(tid * 16);
        const float* src = inputs + rowBase * NF + tid * 4;
        #pragma unroll
        for (int r = 0; r < 5; ++r) {
            asm volatile("global_load_async_to_lds_b128 %0, %1, off"
                         :: "v"(sxOff + r * 4096), "v"(src + r * 1024)
                         : "memory");
        }
    }

    // ---- weight staging (overlaps the async DMA above) -------------------
    for (int i = tid; i < NN * H1 * H2; i += THREADS) sW2h[i] = (_Float16)W2[i];
    for (int i = tid; i < NN * H1; i += THREADS) { sW1[i] = W1[i]; sB1[i] = b1[i]; }
    for (int i = tid; i < NN * H2; i += THREADS) { sB2[i] = b2[i]; sWout[i] = Wout[i]; }
    if (tid < NN) { sBout[tid] = bout[tid]; sMin[tid] = minv[tid]; sMax[tid] = maxv[tid]; }
    if (tid < NC * CATS) sEmb[tid] = cemb[tid];
    if (tid < NC) sObias[tid] = cobias[tid];

    asm volatile("s_wait_asynccnt 0" ::: "memory");
    __syncthreads();

    // ---- categorical features: gather, overwrite input slot in-place -----
    {
        const int row = tid;
        #pragma unroll
        for (int c = 0; c < NC; ++c) {
            int idx = ((int)sX[row * NF + NN + c]) & (CATS - 1);
            sX[row * NF + NN + c] = sEmb[c * CATS + idx] + sObias[c];
        }
    }

    // ---- numeric subnets: h2^T = W2^T (20x40) x h1^T (40x16batch) --------
    const int lane   = tid & 31;
    const int wave   = tid >> 5;
    const int half16 = lane >> 4;
    const int l16    = lane & 15;

    #pragma unroll
    for (int fi = 0; fi < 2; ++fi) {
        const int f = wave + fi * 8;                  // this wave's feature

        // A = W2^T, constant per feature. A[M=j][K=k] = W2[f][k][j].
        const _Float16* wf = &sW2h[f * H1 * H2];
        v16h a00, a01, a10, a11;
        #pragma unroll
        for (int h = 0; h < 16; ++h) {
            int kb = h + half16 * 16;   // 0..31
            int k1 = kb + 32;                              // 32..63, real < 40
            int j0 = l16, j1 = l16 + 16;
            bool kok = (k1 < H1);
            a00[h] = wf[kb * H2 + j0];
            a10[h] = (j1 < H2) ? wf[kb * H2 + j1] : (_Float16)0.0f;
            a01[h] = kok ? wf[k1 * H2 + j0] : (_Float16)0.0f;
            a11[h] = (kok && j1 < H2) ? wf[k1 * H2 + j1] : (_Float16)0.0f;
        }

        // Layer-1 params pre-packed as f16 pairs (B layout: K = h + half16*16).
        h2v w1p0[8], b1p0[8], w1p1[4], b1p1[4];
        #pragma unroll
        for (int p = 0; p < 8; ++p) {
            int k = 2 * p + half16 * 16;
            h2v w; w[0] = (_Float16)sW1[f * H1 + k]; w[1] = (_Float16)sW1[f * H1 + k + 1];
            h2v b; b[0] = (_Float16)sB1[f * H1 + k]; b[1] = (_Float16)sB1[f * H1 + k + 1];
            w1p0[p] = w; b1p0[p] = b;
        }
        #pragma unroll
        for (int p = 0; p < 4; ++p) {                  // K chunk 1: only 32..39 real
            int k = 32 + 2 * p + half16 * 16;
            bool ok = (k < H1);                        // false for upper half-wave
            h2v w = {}, b = {};
            if (ok) {
                w[0] = (_Float16)sW1[f * H1 + k]; w[1] = (_Float16)sW1[f * H1 + k + 1];
                b[0] = (_Float16)sB1[f * H1 + k]; b[1] = (_Float16)sB1[f * H1 + k + 1];
            }
            w1p1[p] = w; b1p1[p] = b;
        }

        // Epilogue constants: this lane's channels j = r + half16*8 (+16 for c1).
        float b2c0[8], wc0[8], b2c1[4], wc1[4];
        #pragma unroll
        for (int r = 0; r < 8; ++r) {
            int j = r + half16 * 8;
            b2c0[r] = sB2[f * H2 + j]; wc0[r] = sWout[f * H2 + j];
        }
        #pragma unroll
        for (int r = 0; r < 4; ++r) {
            int j = 16 + r + half16 * 8;
            bool ok = (j < H2);
            b2c1[r] = ok ? sB2[f * H2 + j] : 0.0f;
            wc1[r]  = ok ? sWout[f * H2 + j] : 0.0f;
        }
        const float lo = sMin[f], hi = sMax[f], ob = sBout[f];

        // B build: h1^T for one 16-row batch tile, packed f16 math.
        auto buildB = [&](int t, v16h& B0, v16h& B1) {
            float x = fminf(fmaxf(sX[(t * 16 + l16) * NF + f], lo), hi);
            _Float16 xh = (_Float16)x;
            h2v xv; xv[0] = xh; xv[1] = xh;
            const h2v zz = {};
            BPack s0, s1;
            #pragma unroll
            for (int p = 0; p < 8; ++p)
                s0.p[p] = __builtin_elementwise_max(
                              __builtin_elementwise_fma(xv, w1p0[p], b1p0[p]), zz);
            #pragma unroll
            for (int p = 0; p < 4; ++p)
                s1.p[p] = __builtin_elementwise_max(
                              __builtin_elementwise_fma(xv, w1p1[p], b1p1[p]), zz);
            #pragma unroll
            for (int p = 4; p < 8; ++p) s1.p[p] = zz;  // K >= 40: pure padding
            B0 = __builtin_bit_cast(v16h, s0);
            B1 = __builtin_bit_cast(v16h, s1);
        };

        // One pipelined step: WMMA(t) -> build B(t+1) -> epilogue(t).
        auto step = [&](int t, const v16h& B0, const v16h& B1, v16h& N0, v16h& N1) {
            v8f c0 = {}, c1 = {};
            c0 = WMMA_F16(a00, B0, c0);
            c0 = WMMA_F16(a01, B1, c0);
            c1 = WMMA_F16(a10, B0, c1);
            c1 = WMMA_F16(a11, B1, c1);

            buildB((t < 15) ? t + 1 : 15, N0, N1);     // fills the WMMA->VALU hazard

            float acc = 0.0f;
            #pragma unroll
            for (int r = 0; r < 8; ++r)
                acc = fmaf(fmaxf(c0[r] + b2c0[r], 0.0f), wc0[r], acc);
            #pragma unroll
            for (int r = 0; r < 4; ++r)
                acc = fmaf(fmaxf(c1[r] + b2c1[r], 0.0f), wc1[r], acc);
            acc += __shfl_xor(acc, 16, 32);            // combine the two half-waves
            // Both half-waves now hold the identical value and target the same
            // address: unconditional store avoids exec-mask juggling.
            sX[(t * 16 + l16) * NF + f] = acc + ob;
        };

        v16h B0a, B1a, B0b, B1b;
        buildB(0, B0a, B1a);
        #pragma unroll 1
        for (int t2 = 0; t2 < 8; ++t2) {               // 16 tiles, ping-pong buffers
            step(2 * t2,     B0a, B1a, B0b, B1b);
            step(2 * t2 + 1, B0b, B1b, B0a, B1a);
        }
    }

    // ---- async DMA of the finished tile from LDS to global ---------------
    __syncthreads();
    {
        const uint32_t sxOff = (uint32_t)(uintptr_t)(&sX[0]) + (uint32_t)(tid * 16);
        float* dst = out + rowBase * NF + tid * 4;
        #pragma unroll
        for (int r = 0; r < 5; ++r) {
            asm volatile("global_store_async_from_lds_b128 %0, %1, off"
                         :: "v"(dst + r * 1024), "v"(sxOff + r * 4096)
                         : "memory");
        }
        asm volatile("s_wait_asynccnt 0" ::: "memory");
    }
}

extern "C" void kernel_launch(void* const* d_in, const int* in_sizes, int n_in,
                              void* d_out, int out_size, void* d_ws, size_t ws_size,
                              hipStream_t stream) {
    const float* inputs = (const float*)d_in[0];
    const float* W1     = (const float*)d_in[1];
    const float* b1     = (const float*)d_in[2];
    const float* W2     = (const float*)d_in[3];
    const float* b2     = (const float*)d_in[4];
    const float* Wout   = (const float*)d_in[5];
    const float* bout   = (const float*)d_in[6];
    const float* minv   = (const float*)d_in[7];
    const float* maxv   = (const float*)d_in[8];
    const float* cemb   = (const float*)d_in[9];
    const float* cobias = (const float*)d_in[10];
    float* outp = (float*)d_out;

    const int rows   = in_sizes[0] / NF;   // 262144
    const int blocks = rows / TILE;        // 1024

    nam_forward_kernel<<<blocks, THREADS, 0, stream>>>(
        inputs, W1, b1, W2, b2, Wout, bout, minv, maxv, cemb, cobias, outp);
}
